// AttentionBlock_73486890435237
// MI455X (gfx1250) — matrix operand order repaired and checked by reading, not compiled, not run on hardware
//
#include <hip/hip_runtime.h>
#include <hip/hip_bf16.h>
#include <math.h>

typedef __attribute__((ext_vector_type(16))) _Float16 v16h;
typedef __attribute__((ext_vector_type(8)))  float    v8f;

#define Bq 2
#define Sq 2048
#define Dq 1024
#define Hq 16
#define HDq 64
#define FFq 4096
#define Mrows (Bq*Sq)   // 4096

// ---------------------------------------------------------------------------
// Fragment loaders (CDNA5 wave32 WMMA layouts, 16-bit A 16x32 / B 32x16)
// A: lanes 0-15 hold M=0..15 with K pairs {0..7,16..23}; lanes 16-31 same M,
//    K pairs {8..15,24..31}  (CDNA5 ISA 7.12.2)
// B: lane&15 = N column; halves hold K {0..7,16..23} (lanes 0-15) / {8..15,24..31} (lanes 16-31)
// C: VGPR v -> M = v + 8*(lane>>4); N = lane&15
// ---------------------------------------------------------------------------
__device__ __forceinline__ v16h load_a_frag(const _Float16* As, int row0, int ldk, int koff) {
  const int lane = threadIdx.x & 31;
  const _Float16* p = As + (size_t)(row0 + (lane & 15)) * ldk + koff;
  const int bk = (lane >> 4) * 8;
  v16h a;
#pragma unroll
  for (int j = 0; j < 8; ++j) {
    const int k0 = (j < 4) ? (bk + 2 * j) : (16 + bk + 2 * (j - 4));
    a[2 * j]     = p[k0];
    a[2 * j + 1] = p[k0 + 1];
  }
  return a;
}

__device__ __forceinline__ v16h load_b_frag(const _Float16* Bs, int n0, int ldn, int koff) {
  const int lane = threadIdx.x & 31;
  const _Float16* p = Bs + (size_t)koff * ldn + n0 + (lane & 15);
  const int bk = (lane >> 4) * 8;
  v16h b;
#pragma unroll
  for (int s = 0; s < 16; ++s) {
    const int k = (s < 8) ? (bk + s) : (16 + bk + (s - 8));
    b[s] = p[(size_t)k * ldn];
  }
  return b;
}

// ---------------------------------------------------------------------------
// f32 -> f16 conversion WITH transpose: out[k*N + n] = in[n*K + k]
// (weights pre-transposed once so GEMM B tiles are contiguous b128 copies)
// ---------------------------------------------------------------------------
__global__ __launch_bounds__(256)
void cvt_t_f16(const float* __restrict__ in, _Float16* __restrict__ out, int N, int K) {
  __shared__ _Float16 tile[32][33];
  const int k0 = blockIdx.x * 32;
  const int n0 = blockIdx.y * 32;
  const int tx = threadIdx.x & 31;
  const int ty = threadIdx.x >> 5;  // 0..7
#pragma unroll
  for (int i = 0; i < 4; ++i)
    tile[ty + 8 * i][tx] = (_Float16)in[(size_t)(n0 + ty + 8 * i) * K + k0 + tx];
  __syncthreads();
#pragma unroll
  for (int i = 0; i < 4; ++i)
    out[(size_t)(k0 + ty + 8 * i) * N + n0 + tx] = tile[tx][ty + 8 * i];
}

// ---------------------------------------------------------------------------
// K-part reshape: Kt[bh][d][s] <- qkv[(b*S+s)*3D + h*192 + 64 + d]
// done once so the attention hot loop has zero scalar transposes
// ---------------------------------------------------------------------------
__global__ __launch_bounds__(256)
void kt_kernel(const _Float16* __restrict__ qkv, _Float16* __restrict__ kt) {
  __shared__ _Float16 tile[32][33];
  const int bh = blockIdx.z;
  const int b = bh >> 4, h = bh & 15;
  const int s0 = blockIdx.x * 32;
  const int d0 = blockIdx.y * 32;
  const int tx = threadIdx.x & 31;
  const int ty = threadIdx.x >> 5;
#pragma unroll
  for (int i = 0; i < 4; ++i)
    tile[ty + 8 * i][tx] =
        qkv[(size_t)(b * Sq + s0 + ty + 8 * i) * (3 * Dq) + h * 3 * HDq + HDq + d0 + tx];
  __syncthreads();
#pragma unroll
  for (int i = 0; i < 4; ++i)
    kt[((size_t)bh * HDq + d0 + ty + 8 * i) * Sq + s0 + tx] = tile[tx][ty + 8 * i];
}

// ---------------------------------------------------------------------------
// LayerNorm: one block per row (D=1024), 256 threads, f16 output
// ---------------------------------------------------------------------------
__global__ __launch_bounds__(256)
void ln_kernel(const float* __restrict__ x, const float* __restrict__ g,
               const float* __restrict__ b, _Float16* __restrict__ out) {
  __shared__ float rs[8], rss[8];
  const int row = blockIdx.x;
  const float* xr = x + (size_t)row * Dq;
  float v[4];
  float s = 0.f, ss = 0.f;
#pragma unroll
  for (int i = 0; i < 4; ++i) {
    v[i] = xr[threadIdx.x + i * 256];
    s += v[i];
    ss += v[i] * v[i];
  }
#pragma unroll
  for (int o = 16; o > 0; o >>= 1) {
    s  += __shfl_xor(s, o, 32);
    ss += __shfl_xor(ss, o, 32);
  }
  if ((threadIdx.x & 31) == 0) { rs[threadIdx.x >> 5] = s; rss[threadIdx.x >> 5] = ss; }
  __syncthreads();
  float tot = 0.f, tot2 = 0.f;
#pragma unroll
  for (int i = 0; i < 8; ++i) { tot += rs[i]; tot2 += rss[i]; }
  const float mean = tot * (1.f / Dq);
  const float var  = tot2 * (1.f / Dq) - mean * mean;
  const float inv  = rsqrtf(var + 1e-5f);
  _Float16* orow = out + (size_t)row * Dq;
#pragma unroll
  for (int i = 0; i < 4; ++i) {
    const int c = threadIdx.x + i * 256;
    orow[c] = (_Float16)((v[i] - mean) * inv * g[c] + b[c]);
  }
}

// ---------------------------------------------------------------------------
// WMMA GEMM:  C[M,N] = A[M,K] * Wt[K,N] + bias (+resid -> f32 out) (GELU opt)
// block tile 128x128, 8 waves each 32x64 (2x4 frags), K-step 64,
// double-buffered LDS (64KB), global_prefetch for the k+2 panel.
// ---------------------------------------------------------------------------
template <bool GELU, bool RESID>
__global__ __launch_bounds__(256)
void gemm_f16(const _Float16* __restrict__ A, const _Float16* __restrict__ Wt,
              const float* __restrict__ bias, const float* __restrict__ resid,
              _Float16* __restrict__ out16, float* __restrict__ out32,
              int N, int K) {
  __shared__ alignas(64) _Float16 Alds[2][128 * 64];
  __shared__ alignas(64) _Float16 Blds[2][64 * 128];

  const int t    = threadIdx.x;
  const int lane = t & 31;
  const int w    = t >> 5;
  const int m0 = blockIdx.y * 128;
  const int n0 = blockIdx.x * 128;
  const int wm = (w & 3) * 32;   // 0,32,64,96
  const int wn = (w >> 2) * 64;  // 0,64

  // tile-fill assignments (32 halves = 4x b128 per thread per tile)
  const int ar = t >> 1;        // 0..127
  const int ac = (t & 1) * 32;  // 0,32
  const int bk = t >> 2;        // 0..63
  const int bc = (t & 3) * 32;  // 0,32,64,96

  v8f acc[2][4] = {};

  auto load_tiles = [&](int k0, int bufi) {
    const uint4* ap = (const uint4*)(A + (size_t)(m0 + ar) * K + k0 + ac);
    uint4* ad = (uint4*)&Alds[bufi][ar * 64 + ac];
    const uint4* bp = (const uint4*)(Wt + (size_t)(k0 + bk) * N + n0 + bc);
    uint4* bd = (uint4*)&Blds[bufi][bk * 128 + bc];
#pragma unroll
    for (int i = 0; i < 4; ++i) ad[i] = ap[i];
#pragma unroll
    for (int i = 0; i < 4; ++i) bd[i] = bp[i];
    if (k0 + 128 < K) {  // prime GL2 for the panel after next
      __builtin_prefetch(A + (size_t)(m0 + ar) * K + k0 + 128 + ac, 0, 1);
      __builtin_prefetch(Wt + (size_t)(k0 + 128 + bk) * N + n0 + bc, 0, 1);
    }
  };

  auto compute = [&](int bufi) {
#pragma unroll
    for (int ks = 0; ks < 64; ks += 32) {
      v16h af[2];
#pragma unroll
      for (int i = 0; i < 2; ++i) af[i] = load_a_frag(&Alds[bufi][0], wm + i * 16, 64, ks);
#pragma unroll
      for (int j = 0; j < 4; ++j) {
        v16h bf = load_b_frag(&Blds[bufi][0], wn + j * 16, 128, ks);
#pragma unroll
        for (int i = 0; i < 2; ++i)
          acc[i][j] = __builtin_amdgcn_wmma_f32_16x16x32_f16(
              false, af[i], false, bf, (short)0, acc[i][j], false, false);
      }
    }
  };

  load_tiles(0, 0);
  __syncthreads();
  int buf = 0;
  for (int k0 = 0; k0 < K; k0 += 64) {
    if (k0 + 64 < K) load_tiles(k0 + 64, buf ^ 1);
    compute(buf);
    __syncthreads();
    buf ^= 1;
  }

  // epilogue
  const int cn = lane & 15;
  const int rb = (lane >> 4) * 8;
#pragma unroll
  for (int i = 0; i < 2; ++i) {
#pragma unroll
    for (int j = 0; j < 4; ++j) {
      const int ncol = n0 + wn + j * 16 + cn;
      const float bv = bias[ncol];
#pragma unroll
      for (int v = 0; v < 8; ++v) {
        const int mrow = m0 + wm + i * 16 + rb + v;
        float val = acc[i][j][v] + bv;
        if (GELU) val = 0.5f * val * (1.0f + erff(val * 0.70710678118f));
        if (RESID) {
          out32[(size_t)mrow * N + ncol] = val + resid[(size_t)mrow * N + ncol];
        } else {
          out16[(size_t)mrow * N + ncol] = (_Float16)val;
        }
      }
    }
  }
}

// ---------------------------------------------------------------------------
// Flash attention: grid (S/64, B*H), 256 threads.
// Q tile resident; K^T streamed from pre-transposed Kt; online softmax.
// ---------------------------------------------------------------------------
__global__ __launch_bounds__(256)
void attn_kernel(const _Float16* __restrict__ qkv, const _Float16* __restrict__ kt,
                 _Float16* __restrict__ vals) {
  __shared__ alignas(64) _Float16 Qs[64 * 64];   // [q][d]
  __shared__ alignas(64) _Float16 Kts[64 * 64];  // [d][k]
  __shared__ alignas(64) _Float16 Vs[64 * 64];   // [k][d]
  __shared__ float    Ss[64 * 64];
  __shared__ alignas(64) _Float16 Ps[64 * 64];
  __shared__ float mrow[64], lrow[64], arow[64];

  const int t    = threadIdx.x;
  const int lane = t & 31;
  const int w    = t >> 5;
  const int bh = blockIdx.y;
  const int b  = bh >> 4;
  const int h  = bh & 15;
  const int q0 = blockIdx.x * 64;
  const size_t rowbase = (size_t)(b * Sq) * (3 * Dq);
  const int qoff = h * 3 * HDq;
  const int voff = qoff + 2 * HDq;
  const _Float16* ktb = kt + (size_t)bh * HDq * Sq;

  for (int idx = t; idx < 512; idx += 256) {
    const int q = idx >> 3, c = (idx & 7) * 8;
    *(uint4*)&Qs[q * 64 + c] =
        *(const uint4*)&qkv[rowbase + (size_t)(q0 + q) * (3 * Dq) + qoff + c];
  }
  if (t < 64) { mrow[t] = -1e30f; lrow[t] = 0.f; }

  const int wq = (w & 3) * 16;
  const int wk = (w >> 2) * 32;
  const int wd = (w >> 2) * 32;

  v8f Of[2] = {};
  __syncthreads();

  for (int kb = 0; kb < Sq; kb += 64) {
    for (int idx = t; idx < 512; idx += 256) {
      const int r = idx >> 3, c = (idx & 7) * 8;
      // K^T tile: rows are d, cols are s (contiguous in kt)
      *(uint4*)&Kts[r * 64 + c] = *(const uint4*)&ktb[(size_t)r * Sq + kb + c];
      *(uint4*)&Vs[r * 64 + c] =
          *(const uint4*)&qkv[rowbase + (size_t)(kb + r) * (3 * Dq) + voff + c];
    }
    __syncthreads();

    v8f sa[2] = {};
#pragma unroll
    for (int dk = 0; dk < 64; dk += 32) {
      v16h a = load_a_frag(Qs, wq, 64, dk);
#pragma unroll
      for (int j = 0; j < 2; ++j) {
        v16h bf = load_b_frag(Kts, wk + j * 16, 64, dk);
        sa[j] = __builtin_amdgcn_wmma_f32_16x16x32_f16(
            false, a, false, bf, (short)0, sa[j], false, false);
      }
    }
    {
      const int cn = lane & 15, rb = (lane >> 4) * 8;
#pragma unroll
      for (int j = 0; j < 2; ++j)
#pragma unroll
        for (int v = 0; v < 8; ++v)
          Ss[(wq + rb + v) * 64 + wk + j * 16 + cn] = sa[j][v] * 0.125f;
    }
    __syncthreads();

    {  // online softmax: 4 threads per row
      const int r = t >> 2, c0 = (t & 3) * 16;
      float sv[16];
      float mx = -1e30f;
#pragma unroll
      for (int i = 0; i < 16; ++i) { sv[i] = Ss[r * 64 + c0 + i]; mx = fmaxf(mx, sv[i]); }
      mx = fmaxf(mx, __shfl_xor(mx, 1, 32));
      mx = fmaxf(mx, __shfl_xor(mx, 2, 32));
      const float mo = mrow[r];
      const float mn = fmaxf(mo, mx);
      float sum = 0.f;
      v16h p;
#pragma unroll
      for (int i = 0; i < 16; ++i) {
        const float e = __expf(sv[i] - mn);
        sum += e;
        p[i] = (_Float16)e;
      }
      *(v16h*)&Ps[r * 64 + c0] = p;
      sum += __shfl_xor(sum, 1, 32);
      sum += __shfl_xor(sum, 2, 32);
      if ((t & 3) == 0) {
        arow[r] = __expf(mo - mn);
        lrow[r] = lrow[r] * arow[r] + sum;
        mrow[r] = mn;
      }
    }
    __syncthreads();

    {  // O = O*alpha + P*V
      const int rb = (lane >> 4) * 8;
#pragma unroll
      for (int j = 0; j < 2; ++j)
#pragma unroll
        for (int v = 0; v < 8; ++v) Of[j][v] *= arow[wq + rb + v];
#pragma unroll
      for (int kk = 0; kk < 64; kk += 32) {
        v16h a = load_a_frag(Ps, wq, 64, kk);
#pragma unroll
        for (int j = 0; j < 2; ++j) {
          v16h bf = load_b_frag(Vs, wd + j * 16, 64, kk);
          Of[j] = __builtin_amdgcn_wmma_f32_16x16x32_f16(
              false, a, false, bf, (short)0, Of[j], false, false);
        }
      }
    }
    __syncthreads();
  }

  {
    const int cn = lane & 15, rb = (lane >> 4) * 8;
#pragma unroll
    for (int j = 0; j < 2; ++j)
#pragma unroll
      for (int v = 0; v < 8; ++v) {
        const int r = wq + rb + v;
        const float o = Of[j][v] / lrow[r];
        const int drow = b * Sq + q0 + r;
        const int dcol = h * HDq + wd + j * 16 + cn;
        vals[(size_t)drow * Dq + dcol] = (_Float16)o;
      }
  }
}

// ---------------------------------------------------------------------------
extern "C" void kernel_launch(void* const* d_in, const int* in_sizes, int n_in,
                              void* d_out, int out_size, void* d_ws, size_t ws_size,
                              hipStream_t stream) {
  const float* x    = (const float*)d_in[0];
  const float* g1   = (const float*)d_in[1];
  const float* b1   = (const float*)d_in[2];
  const float* Wqkv = (const float*)d_in[3];
  const float* bqkv = (const float*)d_in[4];
  const float* Wo   = (const float*)d_in[5];
  const float* bo   = (const float*)d_in[6];
  const float* g2   = (const float*)d_in[7];
  const float* b2   = (const float*)d_in[8];
  const float* W1   = (const float*)d_in[9];
  const float* b1m  = (const float*)d_in[10];
  const float* W2   = (const float*)d_in[11];
  const float* b2m  = (const float*)d_in[12];
  float* out = (float*)d_out;

  char* ws = (char*)d_ws;
  _Float16* h16    = (_Float16*)(ws);                // 8 MB (LN out, reused for LN2)
  _Float16* qkv16  = (_Float16*)(ws + (8u << 20));   // 32 MB (qkv, reused as MLP mid)
  _Float16* m16    = qkv16;
  _Float16* vals16 = (_Float16*)(ws + (40u << 20));  // 8 MB
  float*    x1     = (float*)(ws + (48u << 20));     // 16 MB
  _Float16* wbuf   = (_Float16*)(ws + (64u << 20));  // 8 MB (current weight, f16, K-major)
  _Float16* kt16   = (_Float16*)(ws + (72u << 20));  // 8 MB (K pre-transposed)

  const dim3 blk(256);

  // ---- attention half ----
  cvt_t_f16<<<dim3(Dq / 32, (3 * Dq) / 32), blk, 0, stream>>>(Wqkv, wbuf, 3 * Dq, Dq);
  ln_kernel<<<dim3(Mrows), blk, 0, stream>>>(x, g1, b1, h16);
  gemm_f16<false, false><<<dim3((3 * Dq) / 128, Mrows / 128), blk, 0, stream>>>(
      h16, wbuf, bqkv, nullptr, qkv16, nullptr, 3 * Dq, Dq);
  kt_kernel<<<dim3(Sq / 32, HDq / 32, Bq * Hq), blk, 0, stream>>>(qkv16, kt16);
  attn_kernel<<<dim3(Sq / 64, Bq * Hq), blk, 0, stream>>>(qkv16, kt16, vals16);
  cvt_t_f16<<<dim3(Dq / 32, Dq / 32), blk, 0, stream>>>(Wo, wbuf, Dq, Dq);
  gemm_f16<false, true><<<dim3(Dq / 128, Mrows / 128), blk, 0, stream>>>(
      vals16, wbuf, bo, x, nullptr, x1, Dq, Dq);

  // ---- MLP half ----
  ln_kernel<<<dim3(Mrows), blk, 0, stream>>>(x1, g2, b2, h16);
  cvt_t_f16<<<dim3(Dq / 32, FFq / 32), blk, 0, stream>>>(W1, wbuf, FFq, Dq);
  gemm_f16<true, false><<<dim3(FFq / 128, Mrows / 128), blk, 0, stream>>>(
      h16, wbuf, b1m, nullptr, m16, nullptr, FFq, Dq);
  cvt_t_f16<<<dim3(FFq / 32, Dq / 32), blk, 0, stream>>>(W2, wbuf, Dq, FFq);
  gemm_f16<false, true><<<dim3(Dq / 128, Mrows / 128), blk, 0, stream>>>(
      m16, wbuf, b2m, x1, nullptr, out, Dq, FFq);
}
